// SwinSnake_Alter_49555332661942
// MI455X (gfx1250) — matrix operand order repaired and checked by reading, not compiled, not run on hardware
//
#include <hip/hip_runtime.h>
#include <hip/hip_bf16.h>
#include <math.h>

typedef __attribute__((ext_vector_type(16))) __bf16 v16bf;
typedef __attribute__((ext_vector_type(8)))  float  v8f;

#define SS_ 4

// ---------------- workspace layout (bytes) ----------------
#define WS_FEAT_OFF   0ull
#define WS_FEAT_BYTES (33554432ull)                  /* 8*256*64*64*4  f32 */
#define WS_WQKV_OFF   (WS_FEAT_OFF + WS_FEAT_BYTES)  /* bf16 [768][256]    */
#define WS_WQKV_BYTES (768ull*256ull*2ull)
#define WS_WPROJ_OFF  (WS_WQKV_OFF + WS_WQKV_BYTES)  /* bf16 [256][256]    */
#define WS_WPROJ_BYTES (256ull*256ull*2ull)
#define WS_WDSC_OFF   (WS_WPROJ_OFF + WS_WPROJ_BYTES)/* bf16 [256][2304]   */

static __device__ __forceinline__ __bf16 f2bf(float f) {
  unsigned int u = __builtin_bit_cast(unsigned int, f);
  unsigned int r = (u + 0x7FFFu + ((u >> 16) & 1u)) >> 16;
  unsigned short s = (unsigned short)r;
  return __builtin_bit_cast(__bf16, s);
}

// Async global->LDS copy of 16B per lane (ASYNCcnt-tracked).
// ldsaddr = wave-relative LDS byte offset (low 32 bits of generic LDS ptr).
static __device__ __forceinline__ void async_g2l_b128(unsigned ldsaddr,
                                                      unsigned long long gaddr) {
  asm volatile("global_load_async_to_lds_b128 %0, %1, off"
               :: "v"(ldsaddr), "v"(gaddr) : "memory");
}
static __device__ __forceinline__ void wait_asynccnt0() {
  asm volatile("s_wait_asynccnt 0x0" ::: "memory");
}

// A fragment: 16x32 bf16 tile from row-major [M][K] (ld in elements).
// lane lo = M row; K = hi*16 + e
static __device__ __forceinline__ v16bf load_a_frag(const __bf16* base, int ld, int lane) {
  const int lo = lane & 15, hi = (lane >> 4) & 1;
  const __bf16* r0 = base + lo * ld + hi * 16;
  v16bf a;
#pragma unroll
  for (int e = 0; e < 16; ++e) a[e] = r0[e];
  return a;
}

// B fragment (32x16 KxN) from row-major [N][K]: lane lo = N, K = hi*16 + e
static __device__ __forceinline__ v16bf load_b_nk(const __bf16* base, int ld, int lane) {
  const int lo = lane & 15, hi = (lane >> 4) & 1;
  const __bf16* p = base + lo * ld + hi * 16;
  v16bf b;
#pragma unroll
  for (int e = 0; e < 16; ++e) b[e] = p[e];
  return b;
}

// B fragment (32x16 KxN) from row-major [K][N]
static __device__ __forceinline__ v16bf load_b_kn(const __bf16* base, int ld, int lane) {
  const int lo = lane & 15, hi = (lane >> 4) & 1;
  v16bf b;
#pragma unroll
  for (int e = 0; e < 16; ++e) b[e] = base[(hi * 16 + e) * ld + lo];
  return b;
}

static __device__ __forceinline__ v8f wmma_bf16(v16bf a, v16bf b, v8f c) {
  return __builtin_amdgcn_wmma_f32_16x16x32_bf16(false, a, false, b, (short)0, c, false, false);
}

static __device__ __forceinline__ int shift_region(int v) {
  // rows/cols: [0,56) -> 0 ; [56,60) -> 1 ; [60,64) -> 2
  return v < 56 ? 0 : (v < 60 ? 1 : 2);
}

// ============================================================
// Kernel 0: weight conversion / repack to bf16
// ============================================================
__global__ void __launch_bounds__(256)
prep_weights(const float* __restrict__ qkv_w, const float* __restrict__ proj_w,
             const float* __restrict__ dsc_w,
             __bf16* __restrict__ wqkv, __bf16* __restrict__ wproj,
             __bf16* __restrict__ wdsc) {
  int i = blockIdx.x * 256 + threadIdx.x;
  const int NQ = 768 * 256, NP = 256 * 256, ND = 256 * 2304;
  if (i < NQ) {
    wqkv[i] = f2bf(qkv_w[i]);
  } else if (i < NQ + NP) {
    int j = i - NQ;
    wproj[j] = f2bf(proj_w[j]);
  } else if (i < NQ + NP + ND) {
    int j = i - NQ - NP;
    int co = j / 2304, kidx = j - co * 2304;
    int k = kidx >> 8, ci = kidx & 255;
    // wdsc[co][k*256+ci] = dsc_weight[co][ci][k][0]
    wdsc[j] = f2bf(dsc_w[(co * 256 + ci) * 9 + k]);
  }
}

// ============================================================
// Kernel 1: shifted-window attention, one block per window
// block = 256 threads = 8 waves
// dynamic LDS layout (bytes):
//   QKV bf16 [64][776]  @ 0       (99328)  (first 65536B double as XF32 staging)
//   XOS bf16 [64][264]  @ 99328   (33792)  X during QKV, then O
//   Sf  f32  [64][68]   @ 133120  (17408)
//   Pbf bf16 [64][72]   @ 150528  (9216)   total 159744
// ============================================================
__global__ void __launch_bounds__(256)
swin_attn_kernel(const float* __restrict__ x,
                 const __bf16* __restrict__ wqkv, const float* __restrict__ qkv_b,
                 const __bf16* __restrict__ wproj, const float* __restrict__ proj_b,
                 const float* __restrict__ relb,
                 float* __restrict__ feat) {
  extern __shared__ char smem[];
  __bf16* QKV  = (__bf16*)(smem);
  float*  XF32 = (float*)(smem);           // staging overlay, dead before QKV writes
  __bf16* XOS  = (__bf16*)(smem + 99328);
  float*  Sf   = (float*)(smem + 133120);
  __bf16* Pbf  = (__bf16*)(smem + 150528);
  const int QLD = 776, XLD = 264, SLD = 68, PLD = 72;

  const int tid = threadIdx.x;
  const int lane = tid & 31, wave = tid >> 5;
  const int lo = lane & 15, hi = (lane >> 4) & 1;
  const int b = blockIdx.x >> 6, win = blockIdx.x & 63;
  const int nh = win >> 3, nw = win & 7;

  // ---- 1a) async-stage rolled window X (f32) into LDS ----
  {
    const unsigned ldsbase = (unsigned)(unsigned long long)(uintptr_t)smem;
#pragma unroll
    for (int i = 0; i < 16; ++i) {
      int c = tid + (i << 8);        // 4096 chunks of 16B: 64 tokens x 64 chunks
      int t = c >> 6, q = c & 63;    // token, 16B chunk within its 1KB row
      int r = t >> 3, cc = t & 7;
      int h = ((nh << 3) + r + SS_) & 63;   // roll(-4): xr[i] = x[(i+4)%64]
      int w = ((nw << 3) + cc + SS_) & 63;
      unsigned long long ga = (unsigned long long)(uintptr_t)
          (x + ((((size_t)b * 64 + h) * 64 + w) << 8) + (q << 2));
      async_g2l_b128(ldsbase + (unsigned)(t * 1024 + q * 16), ga);
    }
  }
  wait_asynccnt0();
  __syncthreads();

  // ---- 1b) convert staged X f32 -> bf16 tile ----
  for (int idx = tid; idx < 64 * 256; idx += 256) {
    int t = idx >> 8, ch = idx & 255;
    XOS[t * XLD + ch] = f2bf(XF32[(t << 8) + ch]);
  }
  __syncthreads();

  // ---- 2) QKV = X @ Wqkv^T + b  (M=64, N=768, K=256) ----
  for (int nti = 0; nti < 6; ++nti) {
    const int nt = wave * 6 + nti;  // 48 N-tiles total
    v16bf bw[8];
#pragma unroll
    for (int ks = 0; ks < 8; ++ks)
      bw[ks] = load_b_nk(wqkv + (nt * 16) * 256 + ks * 32, 256, lane);
    const float bias = qkv_b[nt * 16 + lo];
    for (int mt = 0; mt < 4; ++mt) {
      v8f acc = {0.f, 0.f, 0.f, 0.f, 0.f, 0.f, 0.f, 0.f};
#pragma unroll
      for (int ks = 0; ks < 8; ++ks) {
        v16bf a = load_a_frag(XOS + (mt * 16) * XLD + ks * 32, XLD, lane);
        acc = wmma_bf16(a, bw[ks], acc);
      }
#pragma unroll
      for (int r = 0; r < 8; ++r) {
        int M = mt * 16 + r + hi * 8;
        QKV[M * QLD + nt * 16 + lo] = f2bf(acc[r] + bias);
      }
    }
  }
  __syncthreads();

  const float scale = 0.17677669529663687f;  // 32^-0.5

  // ---- 3) attention per head ----
  for (int head = 0; head < 8; ++head) {
    // S = Q*K^T : 16 C-tiles, 2 per wave, inner K = 32 (one WMMA)
#pragma unroll
    for (int tix = 0; tix < 2; ++tix) {
      const int id = wave * 2 + tix;
      const int mt = id >> 2, nt = id & 3;
      v16bf a  = load_a_frag(QKV + (mt * 16) * QLD + head * 32, QLD, lane);
      v16bf bb = load_b_nk (QKV + (nt * 16) * QLD + 256 + head * 32, QLD, lane);
      v8f acc = {0.f, 0.f, 0.f, 0.f, 0.f, 0.f, 0.f, 0.f};
      acc = wmma_bf16(a, bb, acc);
      const int N = nt * 16 + lo;
      const int labN = 3 * shift_region((nh << 3) + (N >> 3)) +
                           shift_region((nw << 3) + (N & 7));
#pragma unroll
      for (int r = 0; r < 8; ++r) {
        int M = mt * 16 + r + hi * 8;
        int labM = 3 * shift_region((nh << 3) + (M >> 3)) +
                       shift_region((nw << 3) + (M & 7));
        float v = acc[r] * scale + relb[(head << 12) + (M << 6) + N];
        if (labM != labN) v -= 100.0f;
        Sf[M * SLD + N] = v;
      }
    }
    __syncthreads();

    // softmax rows (one thread per row)
    if (tid < 64) {
      float mx = -1e30f;
      for (int c2 = 0; c2 < 64; ++c2) mx = fmaxf(mx, Sf[tid * SLD + c2]);
      float sum = 0.f;
      for (int c2 = 0; c2 < 64; ++c2) {
        float e = __expf(Sf[tid * SLD + c2] - mx);
        Sf[tid * SLD + c2] = e;
        sum += e;
      }
      float inv = 1.0f / sum;
      for (int c2 = 0; c2 < 64; ++c2)
        Pbf[tid * PLD + c2] = f2bf(Sf[tid * SLD + c2] * inv);
    }
    __syncthreads();

    // O_head = P @ V : M=64, N=32, K=64 ; one C-tile per wave
    {
      const int mt = wave >> 1, nt = wave & 1;
      v8f acc = {0.f, 0.f, 0.f, 0.f, 0.f, 0.f, 0.f, 0.f};
#pragma unroll
      for (int ks = 0; ks < 2; ++ks) {
        v16bf a  = load_a_frag(Pbf + (mt * 16) * PLD + ks * 32, PLD, lane);
        v16bf bb = load_b_kn (QKV + (ks * 32) * QLD + 512 + head * 32 + nt * 16,
                              QLD, lane);
        acc = wmma_bf16(a, bb, acc);
      }
#pragma unroll
      for (int r = 0; r < 8; ++r) {
        int M = mt * 16 + r + hi * 8;
        XOS[M * XLD + head * 32 + nt * 16 + lo] = f2bf(acc[r]);
      }
    }
    __syncthreads();
  }

  // ---- 4) proj: Y = O @ Wproj^T + b, scatter with inverse roll ----
  for (int nti = 0; nti < 2; ++nti) {
    const int nt = wave * 2 + nti;  // 16 N-tiles
    v16bf bw[8];
#pragma unroll
    for (int ks = 0; ks < 8; ++ks)
      bw[ks] = load_b_nk(wproj + (nt * 16) * 256 + ks * 32, 256, lane);
    const float bias = proj_b[nt * 16 + lo];
    for (int mt = 0; mt < 4; ++mt) {
      v8f acc = {0.f, 0.f, 0.f, 0.f, 0.f, 0.f, 0.f, 0.f};
#pragma unroll
      for (int ks = 0; ks < 8; ++ks) {
        v16bf a = load_a_frag(XOS + (mt * 16) * XLD + ks * 32, XLD, lane);
        acc = wmma_bf16(a, bw[ks], acc);
      }
      const int ch = nt * 16 + lo;
#pragma unroll
      for (int r = 0; r < 8; ++r) {
        int M = mt * 16 + r + hi * 8;
        int h = ((nh << 3) + (M >> 3) + SS_) & 63;  // roll back (+4)
        int w = ((nw << 3) + (M & 7) + SS_) & 63;
        feat[(((b * 256 + ch) * 64 + h) << 6) + w] = acc[r] + bias;
      }
    }
  }
}

// ============================================================
// Kernel 2: snake sampling + (C,C,9,1)/stride(9,1) conv as GEMM
// one block per (b, wd, 32 consecutive hd); 256 thr = 8 waves
// M=32 per block halves L2 traffic on the 1.18MB B-panel vs M=16;
// each B-fragment is loaded once and reused by two WMMAs (2 M-tiles).
// dynamic LDS: At bf16 [32][2320] @0 (148480);
//   y0 @148480 (1152); wy @149632 (1152); xc @150784 (1152) -> 151936
// ============================================================
__global__ void __launch_bounds__(256)
snake_conv_kernel(const float* __restrict__ feat, const float* __restrict__ offset,
                  const __bf16* __restrict__ wdsc, const float* __restrict__ dsc_b,
                  float* __restrict__ out) {
  extern __shared__ char smem[];
  __bf16* At  = (__bf16*)smem;
  int*    y0a = (int*)  (smem + 148480);
  float*  wya = (float*)(smem + 149632);
  int*    xca = (int*)  (smem + 150784);
  const int ALD = 2320;

  const int tid = threadIdx.x;
  const int lane = tid & 31, wave = tid >> 5;
  const int lo = lane & 15, hi = (lane >> 4) & 1;
  const int b = blockIdx.x >> 7, rem = blockIdx.x & 127;
  const int wd = rem >> 1, hd0 = (rem & 1) << 5;

  // ---- snake coordinates for 32 tokens ----
  if (tid < 32) {
    const int hd = hd0 + tid;
    float yo[9];
#pragma unroll
    for (int k = 0; k < 9; ++k)
      yo[k] = offset[((b * 18 + k) * 64 + wd) * 64 + hd];
    float yc[9];
    yc[4] = 0.f;
    yc[3] = yo[3]; yc[2] = yo[2] + yc[3]; yc[1] = yo[1] + yc[2]; yc[0] = yo[0] + yc[1];
    yc[5] = yo[5]; yc[6] = yc[5] + yo[6]; yc[7] = yc[6] + yo[7]; yc[8] = yc[7] + yo[8];
#pragma unroll
    for (int k = 0; k < 9; ++k) {
      float y = (float)wd + yc[k];
      y = fminf(fmaxf(y, 0.f), 63.f);
      float y0f = floorf(y);
      y0a[tid * 9 + k] = (int)y0f;
      wya[tid * 9 + k] = y - y0f;
      int xi = hd + k - 4;                       // x-spread is exact integers -> wx==0
      xca[tid * 9 + k] = xi < 0 ? 0 : (xi > 63 ? 63 : xi);
    }
  }
  __syncthreads();

  // ---- build A tile (32 tokens x 2304 bf16), thread = channel ----
  {
    const int ci = tid;
    const float* fp = feat + (size_t)(b * 256 + ci) * 4096;
    for (int j = 0; j < 32; ++j) {
#pragma unroll
      for (int k = 0; k < 9; ++k) {
        int y0 = y0a[j * 9 + k];
        int y1 = y0 + 1; if (y1 > 63) y1 = 63;
        float wy = wya[j * 9 + k];
        int xc = xca[j * 9 + k];
        float v0 = fp[(y0 << 6) + xc];
        float v1 = fp[(y1 << 6) + xc];
        At[j * ALD + (k << 8) + ci] = f2bf(v0 + wy * (v1 - v0));
      }
    }
  }
  __syncthreads();

  // ---- GEMM: M=32 (2 M-tiles), N=256 (2 N-tiles/wave), K=2304 ----
  for (int nti = 0; nti < 2; ++nti) {
    const int nt = wave * 2 + nti;
    v8f acc0 = {0.f, 0.f, 0.f, 0.f, 0.f, 0.f, 0.f, 0.f};
    v8f acc1 = {0.f, 0.f, 0.f, 0.f, 0.f, 0.f, 0.f, 0.f};
    for (int ks = 0; ks < 72; ++ks) {
      v16bf bb = load_b_nk(wdsc + (size_t)(nt * 16) * 2304 + ks * 32, 2304, lane);
      v16bf a0 = load_a_frag(At + ks * 32, ALD, lane);
      v16bf a1 = load_a_frag(At + 16 * ALD + ks * 32, ALD, lane);
      acc0 = wmma_bf16(a0, bb, acc0);
      acc1 = wmma_bf16(a1, bb, acc1);
    }
    const int co = nt * 16 + lo;
    const float bias = dsc_b[co];
#pragma unroll
    for (int r = 0; r < 8; ++r) {
      int j0 = r + hi * 8;
      float v0 = acc0[r] + bias;
      v0 = v0 > 0.f ? v0 : 0.f;  // relu
      out[((size_t)(b * 256 + co) * 64 + wd) * 64 + hd0 + j0] = v0;
      float v1 = acc1[r] + bias;
      v1 = v1 > 0.f ? v1 : 0.f;
      out[((size_t)(b * 256 + co) * 64 + wd) * 64 + hd0 + 16 + j0] = v1;
    }
  }
}

// ============================================================
extern "C" void kernel_launch(void* const* d_in, const int* in_sizes, int n_in,
                              void* d_out, int out_size, void* d_ws, size_t ws_size,
                              hipStream_t stream) {
  const float* x      = (const float*)d_in[0];
  const float* qkv_w  = (const float*)d_in[1];
  const float* qkv_b  = (const float*)d_in[2];
  const float* proj_w = (const float*)d_in[3];
  const float* proj_b = (const float*)d_in[4];
  const float* relb   = (const float*)d_in[5];
  const float* offset = (const float*)d_in[6];
  const float* dsc_w  = (const float*)d_in[7];
  const float* dsc_b  = (const float*)d_in[8];
  float* out = (float*)d_out;

  char* ws = (char*)d_ws;
  float*  feat  = (float*)(ws + WS_FEAT_OFF);
  __bf16* wqkv  = (__bf16*)(ws + WS_WQKV_OFF);
  __bf16* wproj = (__bf16*)(ws + WS_WPROJ_OFF);
  __bf16* wdsc  = (__bf16*)(ws + WS_WDSC_OFF);

  (void)in_sizes; (void)n_in; (void)out_size; (void)ws_size;

  // 0) weight repack to bf16
  const int total = 768 * 256 + 256 * 256 + 256 * 2304;  // 851968
  prep_weights<<<(total + 255) / 256, 256, 0, stream>>>(qkv_w, proj_w, dsc_w,
                                                        wqkv, wproj, wdsc);
  // 1) shifted-window attention -> feat (B,C,H,W) f32
  swin_attn_kernel<<<512, 256, 159744, stream>>>(x, wqkv, qkv_b, wproj, proj_b,
                                                 relb, feat);
  // 2) snake sampling + dsc conv -> out
  snake_conv_kernel<<<1024, 256, 151936, stream>>>(feat, offset, wdsc, dsc_b, out);
}
